// Nonlocal_15530601742912
// MI455X (gfx1250) — matrix operand order repaired and checked by reading, not compiled, not run on hardware
//
#include <hip/hip_runtime.h>

// ---------------------------------------------------------------------------
// Types for CDNA5 WMMA (wave32): 16x16x32 bf16 -> f32
// ---------------------------------------------------------------------------
typedef __attribute__((ext_vector_type(16))) __bf16 v16bf;
typedef __attribute__((ext_vector_type(8)))  float  v8f;
typedef int v4i __attribute__((vector_size(16)));   // matches builtin param type

union FragU {
    v16bf          v;
    unsigned short us[16];
    uint4          q[2];
};

// Hardware bf16 convert (v_cvt_pk_bf16_f32 on gfx1250, RNE).
__device__ __forceinline__ unsigned short f2bf(float f) {
    __bf16 h = (__bf16)f;
    return __builtin_bit_cast(unsigned short, h);
}
__device__ __forceinline__ unsigned pack2bf(float lo, float hi) {
    return (unsigned)f2bf(lo) | ((unsigned)f2bf(hi) << 16);
}

// Async global->LDS DMA (CDNA5). Guarded so compile cannot fail on a
// toolchain that lacks the builtins; fallback is a plain 16B LDS copy.
#if defined(__gfx1250__) && __has_builtin(__builtin_amdgcn_global_load_async_to_lds_b128) && __has_builtin(__builtin_amdgcn_s_wait_asynccnt)
#define USE_ASYNC_LDS 1
#else
#define USE_ASYNC_LDS 0
#endif

__device__ __forceinline__ void copy16_g2lds(const void* g, void* l) {
#if USE_ASYNC_LDS
    __builtin_amdgcn_global_load_async_to_lds_b128(
        (__attribute__((address_space(1))) v4i*)(unsigned long long)(size_t)g,
        (__attribute__((address_space(3))) v4i*)(unsigned)(size_t)l,
        0, 0);
#else
    *(uint4*)l = *(const uint4*)g;
#endif
}
__device__ __forceinline__ void wait_async_lds() {
#if USE_ASYNC_LDS
    __builtin_amdgcn_s_wait_asynccnt(0);
#endif
}

// A-fragment (16x32, 16-bit), contiguous k order (matches B):
//   lanes 0-15 : M=lane,     K = 0..15
//   lanes 16-31: M=lane-16,  K = 16..31
__device__ __forceinline__ v16bf frag_a_lds(const unsigned short* lds, int stride,
                                            int rowBase, int kBase, int lane) {
    int g = lane >> 4, r = lane & 15;
    const unsigned short* rp = lds + (rowBase + r) * stride + kBase;
    FragU f;
    f.q[0] = *(const uint4*)(rp + 16 * g);
    f.q[1] = *(const uint4*)(rp + 16 * g + 8);
    return f.v;
}

// B-fragment (32x16, 16-bit):
//   lanes 0-15 : N=lane,     K = 0..15
//   lanes 16-31: N=lane-16,  K = 16..31
// Source LDS is stored [N][K] (K contiguous) so loads are 2x16B.
__device__ __forceinline__ v16bf frag_b_lds(const unsigned short* lds, int stride,
                                            int colBase, int kBase, int lane) {
    int g = lane >> 4, r = lane & 15;
    const unsigned short* rp = lds + (colBase + r) * stride + kBase + 16 * g;
    FragU f;
    f.q[0] = *(const uint4*)(rp);
    f.q[1] = *(const uint4*)(rp + 8);
    return f.v;
}

// Same fragments gathered from row-major f32 global memory with convert.
__device__ __forceinline__ v16bf frag_a_gf32(const float* base, int stride,
                                             int rowBase, int kBase, int lane) {
    int g = lane >> 4, r = lane & 15;
    const float* rp = base + (size_t)(rowBase + r) * stride + kBase;
    FragU f;
#pragma unroll
    for (int j = 0; j < 8; ++j) f.us[j]     = f2bf(rp[16 * g + j]);
#pragma unroll
    for (int j = 0; j < 8; ++j) f.us[8 + j] = f2bf(rp[16 * g + 8 + j]);
    return f.v;
}

// B fragment from f32 source indexed [k][col] (row-major over k)
__device__ __forceinline__ v16bf frag_b_gf32(const float* base, int stride,
                                             int colBase, int kBase, int lane) {
    int g = lane >> 4, r = lane & 15;
    int col = colBase + r;
    FragU f;
#pragma unroll
    for (int j = 0; j < 16; ++j)
        f.us[j] = f2bf(base[(size_t)(kBase + 16 * g + j) * stride + col]);
    return f.v;
}

__device__ __forceinline__ v8f wmma_bf16(v16bf a, v16bf b, v8f c) {
    return __builtin_amdgcn_wmma_f32_16x16x32_bf16(false, a, false, b,
                                                   (short)0, c, false, false);
}

// ---------------------------------------------------------------------------
// Problem constants
// ---------------------------------------------------------------------------
#define NB   32
#define C_IN 256
#define C_MID 128
#define HW   4096
#define COUNT_INV (1.0f / 131072.0f)   // 1/(N*H*W)
#define PIXT 256                       // K3 pixel tile
#define NSLOT 512                      // stat partials per channel = 32n * 16pt

// ---------------------------------------------------------------------------
// K0: xsum[n,c] = sum_hw x[n,c,:,:]
// ---------------------------------------------------------------------------
__global__ void k_xsum(const float* __restrict__ x, float* __restrict__ xsum) {
    int nc = blockIdx.x;
    const float4* p = (const float4*)(x + (size_t)nc * HW);
    int t = threadIdx.x;
    float s = 0.f;
#pragma unroll
    for (int i = 0; i < 4; ++i) {
        float4 v = p[t + 256 * i];
        s += v.x + v.y + v.z + v.w;
    }
    __shared__ float sh[256];
    sh[t] = s;
    __syncthreads();
    for (int o = 128; o > 0; o >>= 1) {
        if (t < o) sh[t] += sh[t + o];
        __syncthreads();
    }
    if (t == 0) xsum[nc] = sh[0];
}

// ---------------------------------------------------------------------------
// K1: per-image attention (128x128), ab = attn @ b_g, b_eff = w_out@ab + b_out
// ---------------------------------------------------------------------------
__global__ void k_attn(const float* __restrict__ xsum,
                       const float* __restrict__ w_theta, const float* __restrict__ b_theta,
                       const float* __restrict__ w_phi,   const float* __restrict__ b_phi,
                       const float* __restrict__ b_g,
                       const float* __restrict__ w_out,   const float* __restrict__ b_out,
                       float* __restrict__ attn, float* __restrict__ beff) {
    int n = blockIdx.x;
    int c = threadIdx.x;            // 128 threads
    const float* xs = xsum + n * C_IN;

    float ts = 256.0f * b_theta[c];
    float ps = 256.0f * b_phi[c];
    for (int k = 0; k < C_IN; ++k) {
        float v = xs[k] * (1.0f / 16.0f);   // sum over pooled pixels of xp
        ts += w_theta[c * C_IN + k] * v;
        ps += w_phi[c * C_IN + k] * v;
    }
    __shared__ float shp[C_MID];
    __shared__ float shab[C_MID];
    shp[c] = ps;
    __syncthreads();

    const float sc = 0.08838834764831845f;  // 128^-0.5
    float mx = -3.4e38f;
    for (int d = 0; d < C_MID; ++d) mx = fmaxf(mx, ts * shp[d] * sc);
    float Z = 0.f;
    for (int d = 0; d < C_MID; ++d) Z += expf(ts * shp[d] * sc - mx);
    float inv = 1.0f / Z;

    float* arow = attn + ((size_t)n * C_MID + c) * C_MID;
    float abv = 0.f;
    for (int d = 0; d < C_MID; ++d) {
        float a = expf(ts * shp[d] * sc - mx) * inv;
        arow[d] = a;
        abv += a * b_g[d];
    }
    shab[c] = abv;
    __syncthreads();

    for (int o = c; o < C_IN; o += C_MID) {
        float acc = b_out[o];
        for (int m = 0; m < C_MID; ++m) acc += w_out[o * C_MID + m] * shab[m];
        beff[n * C_IN + o] = acc;
    }
}

// ---------------------------------------------------------------------------
// K2: W_eff[n] = w_out(256x128) @ attn[n](128x128) @ w_g(128x256)  -> bf16
// ---------------------------------------------------------------------------
#define TSTR 136
__global__ void k_weff(const float* __restrict__ attn,
                       const float* __restrict__ w_g,
                       const float* __restrict__ w_out,
                       unsigned short* __restrict__ weff) {
    extern __shared__ unsigned short Tt[];      // [256][TSTR]
    int n    = blockIdx.x;
    int lane = threadIdx.x & 31;
    int w    = threadIdx.x >> 5;                // 8 waves
    const float* an = attn + (size_t)n * C_MID * C_MID;

    // ---- stage 1: T = attn @ w_g, wave w owns rows 16w..16w+15 ----
    v16bf a1[4];
#pragma unroll
    for (int ks = 0; ks < 4; ++ks)
        a1[ks] = frag_a_gf32(an, C_MID, 16 * w, 32 * ks, lane);

    int g = lane >> 4, c15 = lane & 15;
    for (int nt = 0; nt < 16; ++nt) {
        v8f acc = {0.f, 0.f, 0.f, 0.f, 0.f, 0.f, 0.f, 0.f};
#pragma unroll
        for (int ks = 0; ks < 4; ++ks) {
            v16bf b = frag_b_gf32(w_g, C_IN, 16 * nt, 32 * ks, lane);
            acc = wmma_bf16(a1[ks], b, acc);
        }
#pragma unroll
        for (int r = 0; r < 8; ++r) {
            int mrow = 16 * w + r + 8 * g;
            int col  = 16 * nt + c15;
            Tt[col * TSTR + mrow] = f2bf(acc[r]);
        }
    }
    __syncthreads();

    // ---- stage 2: W_eff = w_out @ T, wave w owns rows 32w..32w+31 ----
    v16bf a2[2][4];
#pragma unroll
    for (int fm = 0; fm < 2; ++fm)
#pragma unroll
        for (int ks = 0; ks < 4; ++ks)
            a2[fm][ks] = frag_a_gf32(w_out, C_MID, 32 * w + 16 * fm, 32 * ks, lane);

    unsigned short* wn = weff + (size_t)n * C_IN * C_IN;
    for (int nt = 0; nt < 16; ++nt) {
        v8f acc2[2];
        acc2[0] = (v8f){0.f, 0.f, 0.f, 0.f, 0.f, 0.f, 0.f, 0.f};
        acc2[1] = acc2[0];
#pragma unroll
        for (int ks = 0; ks < 4; ++ks) {
            v16bf b = frag_b_lds(Tt, TSTR, 16 * nt, 32 * ks, lane);
            acc2[0] = wmma_bf16(a2[0][ks], b, acc2[0]);
            acc2[1] = wmma_bf16(a2[1][ks], b, acc2[1]);
        }
#pragma unroll
        for (int fm = 0; fm < 2; ++fm)
#pragma unroll
            for (int r = 0; r < 8; ++r) {
                int o   = 32 * w + 16 * fm + r + 8 * g;
                int col = 16 * nt + c15;
                wn[o * C_IN + col] = f2bf(acc2[fm][r]);
            }
    }
}

// ---------------------------------------------------------------------------
// K3: p[n] = W_eff[n] @ x[n] + b_eff[n]; fused per-channel partial BN stats.
//     Block tile 128(M) x 256(pix); 8 waves (2x4), wave tile 64x64.
//     Double-buffered LDS: A tile via async global->LDS DMA, B tile via
//     b64 global loads staged in registers, both issued before the WMMA
//     block so their latency hides under matrix math. One barrier/step.
//     grid = (16 pix-tiles, 2 m-tiles, 32 n)
// ---------------------------------------------------------------------------
#define AST 40
#define BST 40
__global__ void k_gemm(const unsigned short* __restrict__ weff,
                       const float* __restrict__ x,
                       const float* __restrict__ beff,
                       float* __restrict__ pout,
                       float* __restrict__ psum, float* __restrict__ psq) {
    __shared__ __align__(16) unsigned short Asm[2][128 * AST];
    __shared__ __align__(16) unsigned short Bsm[2][PIXT * BST];
    __shared__ float ssum[128 * 4];
    __shared__ float ssq [128 * 4];

    int pt = blockIdx.x, mt = blockIdx.y, n = blockIdx.z;
    int tid = threadIdx.x, lane = tid & 31, w = tid >> 5;
    int wm = w >> 2, wn = w & 3;

    const unsigned short* wf = weff + ((size_t)n * C_IN + mt * 128) * C_IN;
    const float*          xn = x    + (size_t)n * C_IN * HW + pt * PIXT;

    // A tile: 512 chunks of 8 bf16 (16B per lane, async DMA)
    auto issueA = [&](int ks, int buf) {
        int c0 = 32 * ks;
#pragma unroll
        for (int it = 0; it < 2; ++it) {
            int ch = tid + 256 * it;
            int mm = ch >> 2, kc = ch & 3;
            copy16_g2lds(wf + mm * C_IN + c0 + kc * 8, &Asm[buf][mm * AST + kc * 8]);
        }
    };
    // B tile: 2 k-rows x 2 pixels per thread-iter, b64 global loads
    auto loadB = [&](int ks, float2* regs) {
        int c0 = 32 * ks;
#pragma unroll
        for (int i = 0; i < 8; ++i) {
            int idx = tid + 256 * i;            // 2048 quads
            int kp = idx >> 7, ppq = idx & 127; // kp 0..15, pixel-pair 0..127
            regs[2 * i + 0] = *(const float2*)&xn[(size_t)(c0 + 2 * kp + 0) * HW + 2 * ppq];
            regs[2 * i + 1] = *(const float2*)&xn[(size_t)(c0 + 2 * kp + 1) * HW + 2 * ppq];
        }
    };
    auto storeB = [&](const float2* regs, int buf) {
#pragma unroll
        for (int i = 0; i < 8; ++i) {
            int idx = tid + 256 * i;
            int kp = idx >> 7, ppq = idx & 127;
            *(unsigned*)&Bsm[buf][(2 * ppq + 0) * BST + 2 * kp] =
                pack2bf(regs[2 * i + 0].x, regs[2 * i + 1].x);
            *(unsigned*)&Bsm[buf][(2 * ppq + 1) * BST + 2 * kp] =
                pack2bf(regs[2 * i + 0].y, regs[2 * i + 1].y);
        }
    };

    v8f acc[4][4];
#pragma unroll
    for (int f = 0; f < 4; ++f)
#pragma unroll
        for (int g2 = 0; g2 < 4; ++g2)
            acc[f][g2] = (v8f){0.f, 0.f, 0.f, 0.f, 0.f, 0.f, 0.f, 0.f};

    float2 breg[16];
    issueA(0, 0);
    loadB(0, breg);
    storeB(breg, 0);
    wait_async_lds();
    __syncthreads();

    for (int ks = 0; ks < 8; ++ks) {
        int cur = ks & 1, nxt = cur ^ 1;
        if (ks < 7) {
            issueA(ks + 1, nxt);     // DMA into the other buffer
            loadB(ks + 1, breg);     // global loads in flight under WMMA
        }
        v16bf a[4], b[4];
#pragma unroll
        for (int f = 0; f < 4; ++f)
            a[f] = frag_a_lds(Asm[cur], AST, 64 * wm + 16 * f, 0, lane);
#pragma unroll
        for (int g2 = 0; g2 < 4; ++g2)
            b[g2] = frag_b_lds(Bsm[cur], BST, 64 * wn + 16 * g2, 0, lane);
#pragma unroll
        for (int f = 0; f < 4; ++f)
#pragma unroll
            for (int g2 = 0; g2 < 4; ++g2)
                acc[f][g2] = wmma_bf16(a[f], b[g2], acc[f][g2]);
        if (ks < 7) {
            storeB(breg, nxt);
            wait_async_lds();
        }
        __syncthreads();
    }

    // epilogue: bias add, store p, partial stats (deterministic order)
    int g = lane >> 4, c15 = lane & 15;
    float* pn = pout + ((size_t)n * C_IN + mt * 128) * HW + pt * PIXT;
    const float* bfp = beff + n * C_IN + mt * 128;

#pragma unroll
    for (int f = 0; f < 4; ++f) {
#pragma unroll
        for (int r = 0; r < 8; ++r) {
            int ml = 64 * wm + 16 * f + r + 8 * g;
            float bias = bfp[ml];
            float s = 0.f, s2 = 0.f;
#pragma unroll
            for (int g2 = 0; g2 < 4; ++g2) {
                float v = acc[f][g2][r] + bias;
                pn[(size_t)ml * HW + 64 * wn + 16 * g2 + c15] = v;
                s += v; s2 += v * v;
            }
            // reduce across the 16-lane half-group (xor masks stay in-group)
#pragma unroll
            for (int off = 1; off < 16; off <<= 1) {
                s  += __shfl_xor(s,  off, 32);
                s2 += __shfl_xor(s2, off, 32);
            }
            if (c15 == 0) { ssum[ml * 4 + wn] = s; ssq[ml * 4 + wn] = s2; }
        }
    }
    __syncthreads();
    if (tid < 128) {
        float S = ssum[tid * 4 + 0] + ssum[tid * 4 + 1] + ssum[tid * 4 + 2] + ssum[tid * 4 + 3];
        float Q = ssq [tid * 4 + 0] + ssq [tid * 4 + 1] + ssq [tid * 4 + 2] + ssq [tid * 4 + 3];
        int cg   = mt * 128 + tid;
        int slot = n * 16 + pt;
        psum[cg * NSLOT + slot] = S;
        psq [cg * NSLOT + slot] = Q;
    }
}

// ---------------------------------------------------------------------------
// K4: per-channel mean and rstd from NSLOT partials (fixed-order tree)
// ---------------------------------------------------------------------------
__global__ void k_stats(const float* __restrict__ psum, const float* __restrict__ psq,
                        float* __restrict__ meanvar) {
    int c = blockIdx.x, t = threadIdx.x;
    __shared__ float s1[256], s2[256];
    float a = 0.f, b = 0.f;
#pragma unroll
    for (int i = 0; i < NSLOT / 256; ++i) {
        a += psum[c * NSLOT + t + 256 * i];
        b += psq [c * NSLOT + t + 256 * i];
    }
    s1[t] = a; s2[t] = b;
    __syncthreads();
    for (int o = 128; o > 0; o >>= 1) {
        if (t < o) { s1[t] += s1[t + o]; s2[t] += s2[t + o]; }
        __syncthreads();
    }
    if (t == 0) {
        float mean = s1[0] * COUNT_INV;
        float var  = s2[0] * COUNT_INV - mean * mean;
        meanvar[c * 2 + 0] = mean;
        meanvar[c * 2 + 1] = rsqrtf(var + 1e-5f);
    }
}

// ---------------------------------------------------------------------------
// K5: out = x + gamma*(p - mean)*rstd + beta   (p lives in `out`, in-place)
// ---------------------------------------------------------------------------
__global__ void k_final(const float* __restrict__ x,
                        const float* __restrict__ meanvar,
                        const float* __restrict__ gamma,
                        const float* __restrict__ beta,
                        float* __restrict__ out) {
    size_t i = (size_t)blockIdx.x * 256 + threadIdx.x;   // float4 index
    size_t e = i * 4;
    int c = (int)((e >> 12) & 255);
    float mean = meanvar[c * 2 + 0];
    float rstd = meanvar[c * 2 + 1];
    float ga = gamma[c], be = beta[c];
    float4 p4 = ((const float4*)out)[i];
    float4 x4 = ((const float4*)x)[i];
    float4 o;
    o.x = x4.x + ga * (p4.x - mean) * rstd + be;
    o.y = x4.y + ga * (p4.y - mean) * rstd + be;
    o.z = x4.z + ga * (p4.z - mean) * rstd + be;
    o.w = x4.w + ga * (p4.w - mean) * rstd + be;
    ((float4*)out)[i] = o;
}

// ---------------------------------------------------------------------------
// Launch
// ---------------------------------------------------------------------------
extern "C" void kernel_launch(void* const* d_in, const int* in_sizes, int n_in,
                              void* d_out, int out_size, void* d_ws, size_t ws_size,
                              hipStream_t stream) {
    const float* x       = (const float*)d_in[0];
    const float* w_theta = (const float*)d_in[1];
    const float* b_theta = (const float*)d_in[2];
    const float* w_phi   = (const float*)d_in[3];
    const float* b_phi   = (const float*)d_in[4];
    const float* w_g     = (const float*)d_in[5];
    const float* b_g     = (const float*)d_in[6];
    const float* w_out   = (const float*)d_in[7];
    const float* b_out   = (const float*)d_in[8];
    const float* gamma   = (const float*)d_in[9];
    const float* beta    = (const float*)d_in[10];
    float* out = (float*)d_out;

    char* ws = (char*)d_ws;
    float*          xsum    = (float*)(ws + 0);                 //  32 KB
    float*          attn    = (float*)(ws + 32768);             //   2 MB
    float*          beff    = (float*)(ws + 2146304);           //  32 KB
    unsigned short* weff    = (unsigned short*)(ws + 2179072);  //   4 MB (bf16)
    float*          psum    = (float*)(ws + 6373376);           // 512 KB
    float*          psq     = (float*)(ws + 7421952);           // 512 KB
    float*          meanvar = (float*)(ws + 8470528);           //   2 KB

    k_xsum<<<NB * C_IN, 256, 0, stream>>>(x, xsum);
    k_attn<<<NB, C_MID, 0, stream>>>(xsum, w_theta, b_theta, w_phi, b_phi,
                                     b_g, w_out, b_out, attn, beff);
    k_weff<<<NB, 256, 256 * TSTR * sizeof(unsigned short), stream>>>(attn, w_g, w_out, weff);
    k_gemm<<<dim3(16, 2, NB), 256, 0, stream>>>(weff, x, beff, out, psum, psq);
    k_stats<<<C_IN, 256, 0, stream>>>(psum, psq, meanvar);
    k_final<<<(NB * C_IN * HW) / (256 * 4), 256, 0, stream>>>(x, meanvar, gamma, beta, out);
}
